// LSTMBaseline_13022340841588
// MI455X (gfx1250) — matrix operand order repaired and checked by reading, not compiled, not run on hardware
//
#include <hip/hip_runtime.h>

#define BSZ   8192
#define TT    96
#define PREDN 24
#define HH    64
#define G4H   256
#define LDSW  72              // padded LDS row width in bf16 elems (144B, 16B aligned, conflict-free)
#define WAVES 4
#define BT    (WAVES * 16)    // 64 batch rows per workgroup
#define NTHREADS (WAVES * 32)

typedef __bf16 bf16;
typedef __attribute__((ext_vector_type(16))) __bf16 v16bf;
typedef __attribute__((ext_vector_type(8)))  float  v8f;

union V16U { v16bf v; uint4 q[2]; };

// native v_rcp_f32 / v_exp_f32 based activations (avoid IEEE div expansion)
__device__ __forceinline__ float fastrcp(float x) { return __builtin_amdgcn_rcpf(x); }
__device__ __forceinline__ float sigm(float x)  { return fastrcp(1.0f + __expf(-x)); }
__device__ __forceinline__ float tanh_(float x) { return 2.0f * fastrcp(1.0f + __expf(-2.0f * x)) - 1.0f; }

__device__ __forceinline__ v8f wmma_bf16(v16bf a, v16bf b, v8f c) {
  // v_wmma_f32_16x16x32_bf16  D = A*B + C
  return __builtin_amdgcn_wmma_f32_16x16x32_bf16(false, a, false, b, (short)0, c, false, false);
}

// B operand (32x16 bf16) from LDS weight matrix stored row-major [256][LDSW]
// (row n of W == column n of W^T == B column n). lanes 0-15: K 0-7 & 16-23; lanes 16-31: K 8-15 & 24-31.
__device__ __forceinline__ v16bf load_B(const bf16* __restrict__ W, int lane, int nt, int kt) {
  const bf16* p = W + (nt * 16 + (lane & 15)) * LDSW + kt * 32 + ((lane >> 4) << 3);
  V16U u;
  u.q[0] = *(const uint4*)(p);
  u.q[1] = *(const uint4*)(p + 16);
  return u.v;
}

// A operand (16x32 bf16) from wave-private LDS h tile [16][LDSW].
// lanes 0-15: K 0-7 & 16-23 of row=lane; lanes 16-31: K 8-15 & 24-31 of row=lane-16.
__device__ __forceinline__ v16bf load_Aop(const bf16* __restrict__ buf, int lane, int op) {
  const bf16* p = buf + (lane & 15) * LDSW + ((lane >> 4) << 3) + op * 32;
  V16U u;
  u.q[0] = *(const uint4*)(p);
  u.q[1] = *(const uint4*)(p + 16);
  return u.v;
}

// A operand directly from global bf16 [B][64] row-major (decoder initial state)
__device__ __forceinline__ v16bf load_Aop_g(const bf16* __restrict__ g, int row, int lane, int op) {
  const bf16* p = g + row * HH + ((lane >> 4) << 3) + op * 32;
  V16U u;
  u.q[0] = *(const uint4*)(p);
  u.q[1] = *(const uint4*)(p + 16);
  return u.v;
}

__device__ __forceinline__ void stage_mat(const float* __restrict__ g, bf16* __restrict__ s, int tid) {
  for (int i = tid; i < G4H * HH; i += NTHREADS)
    s[(i >> 6) * LDSW + (i & 63)] = (bf16)g[i];
}

// One LSTM cell step for 16 batch rows (one wave).
// XMAT: input comes from a full [16,64] matrix (A operand aX) through Wih (layer 1).
// !XMAT: scalar input per row, rank-1 term xv[r] * wcol[n] folded into accumulator init.
template <bool XMAT, bool WANTH>
__device__ __forceinline__ void lstm_step(
    const v16bf* aH, const v16bf* aX,
    const bf16* __restrict__ Whh, const bf16* __restrict__ Wih,
    const float* __restrict__ bias, const float* __restrict__ wcol,
    const float* xv,
    float (&c)[4][8],
    bf16* __restrict__ hbuf,
    float (*hout)[8],
    int lane)
{
  const int nl = lane & 15;
  const int hi = (lane >> 4) * 8;
#pragma unroll
  for (int jn = 0; jn < 4; ++jn) {
    v8f acc[4];
#pragma unroll
    for (int g = 0; g < 4; ++g) {            // gate order i, f, g, o
      const int n = g * 64 + jn * 16 + nl;
      const float b = bias[n];
      v8f a;
      if (XMAT) {
#pragma unroll
        for (int r = 0; r < 8; ++r) a[r] = b;
      } else {
        const float w = wcol[n];
#pragma unroll
        for (int r = 0; r < 8; ++r) a[r] = b + xv[r] * w;
      }
      const int nt = g * 4 + jn;
      a = wmma_bf16(aH[0], load_B(Whh, lane, nt, 0), a);
      a = wmma_bf16(aH[1], load_B(Whh, lane, nt, 1), a);
      if (XMAT) {
        a = wmma_bf16(aX[0], load_B(Wih, lane, nt, 0), a);
        a = wmma_bf16(aX[1], load_B(Wih, lane, nt, 1), a);
      }
      acc[g] = a;
    }
#pragma unroll
    for (int r = 0; r < 8; ++r) {
      float iv = sigm(acc[0][r]);
      float fv = sigm(acc[1][r]);
      float gv = tanh_(acc[2][r]);
      float ov = sigm(acc[3][r]);
      float cn = fv * c[jn][r] + iv * gv;
      c[jn][r] = cn;
      float hn = ov * tanh_(cn);
      if (WANTH) hout[jn][r] = hn;
      hbuf[(hi + r) * LDSW + jn * 16 + nl] = (bf16)hn;   // D-layout -> row-major bf16 tile
    }
  }
}

extern "C" __global__ __launch_bounds__(NTHREADS)
void lstm_enc(const float* __restrict__ x,
              const float* __restrict__ Wih0, const float* __restrict__ Whh0, const float* __restrict__ b0,
              const float* __restrict__ Wih1, const float* __restrict__ Whh1, const float* __restrict__ b1,
              bf16* __restrict__ h0T, bf16* __restrict__ h1T,
              float* __restrict__ c0T, float* __restrict__ c1T)
{
  __shared__ bf16  sWhh0[G4H * LDSW];
  __shared__ bf16  sWih1[G4H * LDSW];
  __shared__ bf16  sWhh1[G4H * LDSW];
  __shared__ float sB0[G4H];
  __shared__ float sB1[G4H];
  __shared__ float sW0c[G4H];
  __shared__ bf16  sConv[WAVES * 2 * 16 * LDSW];

  const int tid = threadIdx.x, lane = tid & 31, wave = tid >> 5;
  stage_mat(Whh0, sWhh0, tid);
  stage_mat(Wih1, sWih1, tid);
  stage_mat(Whh1, sWhh1, tid);
  for (int i = tid; i < G4H; i += NTHREADS) { sB0[i] = b0[i]; sB1[i] = b1[i]; sW0c[i] = Wih0[i]; }
  __syncthreads();

  bf16* buf0 = sConv + (wave * 2 + 0) * 16 * LDSW;
  bf16* buf1 = sConv + (wave * 2 + 1) * 16 * LDSW;
  const int rowbase = blockIdx.x * BT + wave * 16;
  const int hi = (lane >> 4) * 8;

  float c0[4][8], c1[4][8];
#pragma unroll
  for (int jn = 0; jn < 4; ++jn)
#pragma unroll
    for (int r = 0; r < 8; ++r) { c0[jn][r] = 0.f; c1[jn][r] = 0.f; }

  v16bf a0[2], a1[2];
  { V16U z; z.q[0] = make_uint4(0, 0, 0, 0); z.q[1] = make_uint4(0, 0, 0, 0);
    a0[0] = z.v; a0[1] = z.v; a1[0] = z.v; a1[1] = z.v; }

  for (int t = 0; t < TT; ++t) {
    float xv[8];
#pragma unroll
    for (int r = 0; r < 8; ++r) xv[r] = x[(rowbase + hi + r) * TT + t];
    lstm_step<false, false>(a0, a0, sWhh0, nullptr, sB0, sW0c, xv, c0, buf0, nullptr, lane);
    a0[0] = load_Aop(buf0, lane, 0); a0[1] = load_Aop(buf0, lane, 1);
    lstm_step<true, false>(a1, a0, sWhh1, sWih1, sB1, nullptr, nullptr, c1, buf1, nullptr, lane);
    a1[0] = load_Aop(buf1, lane, 0); a1[1] = load_Aop(buf1, lane, 1);
  }

  // export final states: h as bf16 row-major [B][64] (A-operand regs are lane-contiguous chunks)
  const int row = rowbase + (lane & 15);
#pragma unroll
  for (int op = 0; op < 2; ++op) {
    V16U u; u.v = a0[op];
    bf16* p = h0T + row * HH + ((lane >> 4) << 3) + op * 32;
    *(uint4*)p = u.q[0]; *(uint4*)(p + 16) = u.q[1];
    u.v = a1[op];
    p = h1T + row * HH + ((lane >> 4) << 3) + op * 32;
    *(uint4*)p = u.q[0]; *(uint4*)(p + 16) = u.q[1];
  }
#pragma unroll
  for (int jn = 0; jn < 4; ++jn)
#pragma unroll
    for (int r = 0; r < 8; ++r) {
      const int m = rowbase + hi + r;
      c0T[m * HH + jn * 16 + (lane & 15)] = c0[jn][r];
      c1T[m * HH + jn * 16 + (lane & 15)] = c1[jn][r];
    }
}

extern "C" __global__ __launch_bounds__(NTHREADS)
void lstm_dec(const float* __restrict__ Wih0, const float* __restrict__ Whh0, const float* __restrict__ b0,
              const float* __restrict__ Wih1, const float* __restrict__ Whh1, const float* __restrict__ b1,
              const float* __restrict__ outW, const float* __restrict__ outb, const float* __restrict__ dstart,
              const bf16* __restrict__ h0T, const bf16* __restrict__ h1T,
              const float* __restrict__ c0T, const float* __restrict__ c1T,
              float* __restrict__ out)
{
  __shared__ bf16  sWhh0[G4H * LDSW];
  __shared__ bf16  sWih1[G4H * LDSW];
  __shared__ bf16  sWhh1[G4H * LDSW];
  __shared__ float sB0[G4H];
  __shared__ float sB1[G4H];
  __shared__ float sW0c[G4H];
  __shared__ float sOW[HH];
  __shared__ bf16  sConv[WAVES * 2 * 16 * LDSW];

  const int tid = threadIdx.x, lane = tid & 31, wave = tid >> 5;
  stage_mat(Whh0, sWhh0, tid);
  stage_mat(Wih1, sWih1, tid);
  stage_mat(Whh1, sWhh1, tid);
  for (int i = tid; i < G4H; i += NTHREADS) { sB0[i] = b0[i]; sB1[i] = b1[i]; sW0c[i] = Wih0[i]; }
  for (int i = tid; i < HH; i += NTHREADS) sOW[i] = outW[i];
  __syncthreads();

  bf16* buf0 = sConv + (wave * 2 + 0) * 16 * LDSW;
  bf16* buf1 = sConv + (wave * 2 + 1) * 16 * LDSW;
  const int rowbase = blockIdx.x * BT + wave * 16;
  const int hi = (lane >> 4) * 8;
  const int row = rowbase + (lane & 15);

  float c0[4][8], c1[4][8];
#pragma unroll
  for (int jn = 0; jn < 4; ++jn)
#pragma unroll
    for (int r = 0; r < 8; ++r) {
      const int m = rowbase + hi + r;
      c0[jn][r] = c0T[m * HH + jn * 16 + (lane & 15)];
      c1[jn][r] = c1T[m * HH + jn * 16 + (lane & 15)];
    }

  v16bf a0[2], a1[2];
  a0[0] = load_Aop_g(h0T, row, lane, 0); a0[1] = load_Aop_g(h0T, row, lane, 1);
  a1[0] = load_Aop_g(h1T, row, lane, 0); a1[1] = load_Aop_g(h1T, row, lane, 1);

  const float st = dstart[0];
  const float ob = outb[0];
  float inp[8];
#pragma unroll
  for (int r = 0; r < 8; ++r) inp[r] = st;

  for (int p = 0; p < PREDN; ++p) {
    float h1f[4][8];
    lstm_step<false, false>(a0, a0, sWhh0, nullptr, sB0, sW0c, inp, c0, buf0, nullptr, lane);
    a0[0] = load_Aop(buf0, lane, 0); a0[1] = load_Aop(buf0, lane, 1);
    lstm_step<true, true>(a1, a0, sWhh1, sWih1, sB1, nullptr, nullptr, c1, buf1, h1f, lane);
    a1[0] = load_Aop(buf1, lane, 0); a1[1] = load_Aop(buf1, lane, 1);

    // output projection: out[m] = h1[m,:] . outW + outb ; reduce across 16 lanes (n dim)
    float part[8];
#pragma unroll
    for (int r = 0; r < 8; ++r) {
      float s = 0.f;
#pragma unroll
      for (int jn = 0; jn < 4; ++jn) s += h1f[jn][r] * sOW[jn * 16 + (lane & 15)];
      part[r] = s;
    }
#pragma unroll
    for (int m = 1; m <= 8; m <<= 1)
#pragma unroll
      for (int r = 0; r < 8; ++r) part[r] += __shfl_xor(part[r], m, 32);
#pragma unroll
    for (int r = 0; r < 8; ++r) {
      part[r] += ob;
      if ((lane & 15) == r) out[(rowbase + hi + r) * PREDN + p] = part[r];
      inp[r] = part[r];   // autoregressive feedback (broadcast across half-wave already)
    }
  }
}

extern "C" void kernel_launch(void* const* d_in, const int* in_sizes, int n_in,
                              void* d_out, int out_size, void* d_ws, size_t ws_size,
                              hipStream_t stream) {
  (void)in_sizes; (void)n_in; (void)out_size; (void)ws_size;
  const float* x      = (const float*)d_in[0];
  const float* eWih0  = (const float*)d_in[1];
  const float* eWhh0  = (const float*)d_in[2];
  const float* eb0    = (const float*)d_in[3];
  const float* eWih1  = (const float*)d_in[4];
  const float* eWhh1  = (const float*)d_in[5];
  const float* eb1    = (const float*)d_in[6];
  const float* dWih0  = (const float*)d_in[7];
  const float* dWhh0  = (const float*)d_in[8];
  const float* db0    = (const float*)d_in[9];
  const float* dWih1  = (const float*)d_in[10];
  const float* dWhh1  = (const float*)d_in[11];
  const float* db1    = (const float*)d_in[12];
  const float* outW   = (const float*)d_in[13];
  const float* outb   = (const float*)d_in[14];
  const float* dstart = (const float*)d_in[15];

  // workspace: h0T,h1T bf16 [B][64]; c0T,c1T f32 [B][64]  (total 6 MB)
  bf16*  h0T = (bf16*)d_ws;
  bf16*  h1T = h0T + (size_t)BSZ * HH;
  float* c0T = (float*)((char*)d_ws + (size_t)2 * BSZ * HH * sizeof(bf16));
  float* c1T = c0T + (size_t)BSZ * HH;

  dim3 grid(BSZ / BT), block(NTHREADS);
  lstm_enc<<<grid, block, 0, stream>>>(x, eWih0, eWhh0, eb0, eWih1, eWhh1, eb1,
                                       h0T, h1T, c0T, c1T);
  lstm_dec<<<grid, block, 0, stream>>>(dWih0, dWhh0, db0, dWih1, dWhh1, db1,
                                       outW, outb, dstart, h0T, h1T, c0T, c1T,
                                       (float*)d_out);
}
